// Encoder_63273458205283
// MI455X (gfx1250) — matrix operand order repaired and checked by reading, not compiled, not run on hardware
//
#include <hip/hip_runtime.h>
#include <math.h>

#define NNODES 50000
#define NEDGES 800000
#define ETOT   (NEDGES + NNODES)   // with self loops
#define FIN    128
#define NHEAD  2
#define C1D    128
#define HC1    (NHEAD * C1D)       // 256
#define C2D    64
#define HC2    (NHEAD * C2D)       // 128
#define NEG_SLOPE 0.2f

typedef __bf16 bf16;
typedef bf16  v16bf __attribute__((ext_vector_type(16)));
typedef float v8f   __attribute__((ext_vector_type(8)));

union Frag16 { v16bf v; unsigned int u[8]; };

__device__ __forceinline__ unsigned short f32_to_bf16_rne(float f) {
  unsigned int u = __float_as_uint(f);
  u += 0x7FFFu + ((u >> 16) & 1u);   // round-to-nearest-even
  return (unsigned short)(u >> 16);
}

// ---------------- elementwise utility kernels ----------------
__global__ void fill_f32(float* __restrict__ p, float v, int n) {
  int i = blockIdx.x * blockDim.x + threadIdx.x;
  if (i < n) p[i] = v;
}

__global__ void cvt_f32_to_bf16(const float* __restrict__ in,
                                unsigned short* __restrict__ out, int n) {
  int i = blockIdx.x * blockDim.x + threadIdx.x;
  if (i < n) out[i] = f32_to_bf16_rne(in[i]);
}

__global__ void copy_f32(const float* __restrict__ in, float* __restrict__ out, int n) {
  int i = blockIdx.x * blockDim.x + threadIdx.x;
  if (i < n) out[i] = in[i];
}

// ---------------- WMMA bf16 GEMM:  C[M,NOUT] = A[M,K] * B[K,NOUT] ----------------
// One wave per 16x16 output tile. M multiple of 16, K multiple of 32, NOUT multiple of 16.
template <int K, int NOUT>
__global__ __launch_bounds__(256)
void gemm_wmma_bf16(const unsigned short* __restrict__ A,
                    const unsigned short* __restrict__ B,
                    float* __restrict__ C, int M) {
  const int NT    = NOUT / 16;
  const int wave  = blockIdx.x * (blockDim.x >> 5) + (threadIdx.x >> 5);
  const int lane  = threadIdx.x & 31;
  const int tiles = (M / 16) * NT;
  if (wave >= tiles) return;                 // wave-uniform: EXEC stays all-ones for WMMA
  const int tm = wave / NT, tn = wave % NT;
  const int row0 = tm * 16, col0 = tn * 16;

  // A-fragment addressing (16-bit A 16x32 layout, ISA 7.12.2)
  const int m     = lane & 15;
  const int kgrpA = (lane < 16) ? 0 : 8;

  v8f acc = {};
#pragma unroll
  for (int k0 = 0; k0 < K; k0 += 32) {
    Frag16 a;
    const unsigned short* Arow = A + (size_t)(row0 + m) * K + k0;
#pragma unroll
    for (int v = 0; v < 8; ++v) {
      const int kk = kgrpA + 2 * v + ((v >= 4) ? 8 : 0);  // {0,2,4,6,16,18,20,22}+kgrp
      a.u[v] = *(const unsigned int*)(Arow + kk);         // pair of bf16
    }
    Frag16 b;
    const unsigned short* Bcol = B + (size_t)k0 * NOUT + col0 + m; // lane = N column, k as A
#pragma unroll
    for (int v = 0; v < 8; ++v) {
      const int kk = kgrpA + 2 * v + ((v >= 4) ? 8 : 0);
      b.u[v] = (unsigned int)Bcol[(size_t)kk * NOUT] | ((unsigned int)Bcol[(size_t)(kk + 1) * NOUT] << 16);
    }

    acc = __builtin_amdgcn_wmma_f32_16x16x32_bf16(
        /*neg_a=*/false, a.v, /*neg_b=*/false, b.v,
        /*c_mod=*/(short)0, acc, /*reuse_a=*/false, /*reuse_b=*/false);
  }

  // C/D layout: lane holds column n=lane&15; VGPR r holds row r (+8 for lanes 16-31)
  const int n     = lane & 15;
  const int mbase = (lane < 16) ? 0 : 8;
#pragma unroll
  for (int r = 0; r < 8; ++r)
    C[(size_t)(row0 + mbase + r) * NOUT + col0 + n] = acc[r];
}

// ---------------- attention alpha dots: as/ad[n,h] = <h[n,h,:], a_{src,dst}[h,:]> ----
__global__ void attn_alpha(const float* __restrict__ h,
                           const float* __restrict__ a_src,
                           const float* __restrict__ a_dst,
                           float* __restrict__ as_, float* __restrict__ ad_, int C) {
  int i = blockIdx.x * blockDim.x + threadIdx.x;       // over NNODES*NHEAD
  if (i >= NNODES * NHEAD) return;
  const int n = i / NHEAD, hd = i % NHEAD;
  const float* hp = h + (size_t)n * NHEAD * C + hd * C;
  const float* sp = a_src + hd * C;
  const float* dp = a_dst + hd * C;
  float accs = 0.f, accd = 0.f;
  for (int c = 0; c < C; ++c) { float v = hp[c]; accs += v * sp[c]; accd += v * dp[c]; }
  as_[i] = accs; ad_[i] = accd;
}

// ---------------- edge helpers ----------------
__device__ __forceinline__ void edge_endpoints(const int* __restrict__ ei, int eid,
                                               int& s, int& d) {
  if (eid < NEDGES) { s = ei[eid]; d = ei[NEDGES + eid]; }
  else              { s = d = eid - NEDGES; }            // self loop
}

__device__ __forceinline__ void atomicMaxFloat(float* addr, float v) {
  if (v >= 0.f) atomicMax((int*)addr, __float_as_int(v));
  else          atomicMin((unsigned int*)addr, __float_as_uint(v));
}

__global__ void edge_logits_max(const int* __restrict__ ei,
                                const float* __restrict__ as_,
                                const float* __restrict__ ad_,
                                float* __restrict__ mmax, float* __restrict__ ebuf) {
  int i = blockIdx.x * blockDim.x + threadIdx.x;       // over ETOT*NHEAD
  if (i >= ETOT * NHEAD) return;
  const int eid = i / NHEAD, hd = i % NHEAD;
  int s, d; edge_endpoints(ei, eid, s, d);
  float e = as_[s * NHEAD + hd] + ad_[d * NHEAD + hd];
  e = (e > 0.f) ? e : NEG_SLOPE * e;                   // leaky relu
  ebuf[i] = e;
  atomicMaxFloat(&mmax[d * NHEAD + hd], e);
}

__global__ void edge_exp_sum(const int* __restrict__ ei,
                             float* __restrict__ ebuf,
                             const float* __restrict__ mmax,
                             float* __restrict__ ssum) {
  int i = blockIdx.x * blockDim.x + threadIdx.x;       // over ETOT*NHEAD
  if (i >= ETOT * NHEAD) return;
  const int eid = i / NHEAD, hd = i % NHEAD;
  int s, d; edge_endpoints(ei, eid, s, d);
  float ex = __expf(ebuf[i] - mmax[d * NHEAD + hd]);
  ebuf[i] = ex;
  atomicAdd(&ssum[d * NHEAD + hd], ex);
}

__global__ void edge_scatter(const int* __restrict__ ei,
                             const float* __restrict__ ebuf,
                             const float* __restrict__ ssum,
                             const float* __restrict__ h,
                             float* __restrict__ out, int C) {
  const int cq = C >> 2;                               // float4 chunks per head
  const int total = ETOT * NHEAD * cq;
  int i = blockIdx.x * blockDim.x + threadIdx.x;
  if (i >= total) return;
  const int c4  = i % cq;
  const int t   = i / cq;
  const int hd  = t % NHEAD;
  const int eid = t / NHEAD;
  int s, d; edge_endpoints(ei, eid, s, d);
  const float alpha = ebuf[t] / (ssum[d * NHEAD + hd] + 1e-16f);
  const float4 hv = *(const float4*)(h + (size_t)s * NHEAD * C + hd * C + (c4 << 2));
  float* op = out + (size_t)d * NHEAD * C + hd * C + (c4 << 2);
  atomicAdd(op + 0, hv.x * alpha);
  atomicAdd(op + 1, hv.y * alpha);
  atomicAdd(op + 2, hv.z * alpha);
  atomicAdd(op + 3, hv.w * alpha);
}

// ---------------- bias + ELU, optional f32 and/or bf16 outputs ----------------
__global__ void finalize_elu(const float* __restrict__ acc, const float* __restrict__ bias,
                             float* __restrict__ out_f32,
                             unsigned short* __restrict__ out_bf16, int HC) {
  int i = blockIdx.x * blockDim.x + threadIdx.x;       // over NNODES*HC
  if (i >= NNODES * HC) return;
  float v = acc[i] + bias[i % HC];
  v = (v > 0.f) ? v : expm1f(v);                       // ELU
  if (out_f32)  out_f32[i]  = v;
  if (out_bf16) out_bf16[i] = f32_to_bf16_rne(v);
}

// ---------------- launch ----------------
static inline int cdiv(long long a, long long b) { return (int)((a + b - 1) / b); }

extern "C" void kernel_launch(void* const* d_in, const int* in_sizes, int n_in,
                              void* d_out, int out_size, void* d_ws, size_t ws_size,
                              hipStream_t stream) {
  const float* x        = (const float*)d_in[0];
  const int*   ei       = (const int*)  d_in[1];
  const float* W1       = (const float*)d_in[2];
  const float* att_src1 = (const float*)d_in[3];
  const float* att_dst1 = (const float*)d_in[4];
  const float* b1       = (const float*)d_in[5];
  const float* W2       = (const float*)d_in[6];
  const float* att_src2 = (const float*)d_in[7];
  const float* att_dst2 = (const float*)d_in[8];
  const float* b2       = (const float*)d_in[9];
  float* out = (float*)d_out;                 // [N*HC2 h2 | N*FIN encoded]

  // workspace carve-up
  char* ws = (char*)d_ws;
  size_t off = 0;
  auto take = [&](size_t bytes) -> void* {
    void* p = ws + off;
    off += (bytes + 255) & ~(size_t)255;
    return p;
  };
  unsigned short* xb   = (unsigned short*)take((size_t)NNODES * FIN * 2);
  unsigned short* w1b  = (unsigned short*)take((size_t)FIN * HC1 * 2);
  unsigned short* w2b  = (unsigned short*)take((size_t)HC1 * HC2 * 2);
  float* h_buf         = (float*)take((size_t)NNODES * HC1 * 4);  // reused layer 2
  float* out_buf       = (float*)take((size_t)NNODES * HC1 * 4);  // reused layer 2
  unsigned short* h1b  = (unsigned short*)take((size_t)NNODES * HC1 * 2);
  float* as_           = (float*)take((size_t)NNODES * NHEAD * 4);
  float* ad_           = (float*)take((size_t)NNODES * NHEAD * 4);
  float* mmax          = (float*)take((size_t)NNODES * NHEAD * 4);
  float* ssum          = (float*)take((size_t)NNODES * NHEAD * 4);
  float* ebuf          = (float*)take((size_t)ETOT * NHEAD * 4);

  const int T = 256;

  // encoded output = x (second tuple element)
  copy_f32<<<cdiv((long long)NNODES * FIN, T), T, 0, stream>>>(x, out + (size_t)NNODES * HC2,
                                                               NNODES * FIN);
  // bf16 conversions for WMMA
  cvt_f32_to_bf16<<<cdiv((long long)NNODES * FIN, T), T, 0, stream>>>(x, xb, NNODES * FIN);
  cvt_f32_to_bf16<<<cdiv(FIN * HC1, T), T, 0, stream>>>(W1, w1b, FIN * HC1);
  cvt_f32_to_bf16<<<cdiv(HC1 * HC2, T), T, 0, stream>>>(W2, w2b, HC1 * HC2);

  // ================= layer 1 (C=128, HC=256) =================
  {
    const int tiles = (NNODES / 16) * (HC1 / 16);     // 50000 tiles, 8 waves/block
    gemm_wmma_bf16<FIN, HC1><<<tiles / 8, T, 0, stream>>>(xb, w1b, h_buf, NNODES);

    attn_alpha<<<cdiv(NNODES * NHEAD, T), T, 0, stream>>>(h_buf, att_src1, att_dst1,
                                                          as_, ad_, C1D);
    fill_f32<<<cdiv(NNODES * NHEAD, T), T, 0, stream>>>(mmax, -INFINITY, NNODES * NHEAD);
    fill_f32<<<cdiv(NNODES * NHEAD, T), T, 0, stream>>>(ssum, 0.f, NNODES * NHEAD);
    fill_f32<<<cdiv((long long)NNODES * HC1, T), T, 0, stream>>>(out_buf, 0.f, NNODES * HC1);

    edge_logits_max<<<cdiv((long long)ETOT * NHEAD, T), T, 0, stream>>>(ei, as_, ad_, mmax, ebuf);
    edge_exp_sum  <<<cdiv((long long)ETOT * NHEAD, T), T, 0, stream>>>(ei, ebuf, mmax, ssum);
    edge_scatter  <<<cdiv((long long)ETOT * NHEAD * (C1D / 4), T), T, 0, stream>>>(
        ei, ebuf, ssum, h_buf, out_buf, C1D);

    // ELU; emit bf16 h1 as layer-2 GEMM input (f32 copy not needed elsewhere)
    finalize_elu<<<cdiv((long long)NNODES * HC1, T), T, 0, stream>>>(out_buf, b1, nullptr,
                                                                    h1b, HC1);
  }

  // ================= layer 2 (C=64, HC=128) =================
  {
    const int tiles = (NNODES / 16) * (HC2 / 16);     // 25000 tiles
    gemm_wmma_bf16<HC1, HC2><<<tiles / 8, T, 0, stream>>>(h1b, w2b, h_buf, NNODES);

    attn_alpha<<<cdiv(NNODES * NHEAD, T), T, 0, stream>>>(h_buf, att_src2, att_dst2,
                                                          as_, ad_, C2D);
    fill_f32<<<cdiv(NNODES * NHEAD, T), T, 0, stream>>>(mmax, -INFINITY, NNODES * NHEAD);
    fill_f32<<<cdiv(NNODES * NHEAD, T), T, 0, stream>>>(ssum, 0.f, NNODES * NHEAD);
    fill_f32<<<cdiv((long long)NNODES * HC2, T), T, 0, stream>>>(out_buf, 0.f, NNODES * HC2);

    edge_logits_max<<<cdiv((long long)ETOT * NHEAD, T), T, 0, stream>>>(ei, as_, ad_, mmax, ebuf);
    edge_exp_sum  <<<cdiv((long long)ETOT * NHEAD, T), T, 0, stream>>>(ei, ebuf, mmax, ssum);
    edge_scatter  <<<cdiv((long long)ETOT * NHEAD * (C2D / 4), T), T, 0, stream>>>(
        ei, ebuf, ssum, h_buf, out_buf, C2D);

    finalize_elu<<<cdiv((long long)NNODES * HC2, T), T, 0, stream>>>(out_buf, b2, out,
                                                                    nullptr, HC2);
  }
}
